// TransformerBlock_24300924961360
// MI455X (gfx1250) — matrix operand order repaired and checked by reading, not compiled, not run on hardware
//
#include <hip/hip_runtime.h>
#include <hip/hip_bf16.h>

typedef __attribute__((ext_vector_type(16))) _Float16 v16h;
typedef __attribute__((ext_vector_type(8)))  _Float16 v8h;
typedef __attribute__((ext_vector_type(8)))  float    v8f;
typedef __attribute__((ext_vector_type(4)))  int      v4i;
typedef _Float16 half_t;

constexpr int cB   = 4;
constexpr int cT   = 2048;
constexpr int cD   = 2048;
constexpr int cG   = 4;
constexpr int cHPG = 4;
constexpr int cHD  = 128;
constexpr int cGHD = cG * cHD;        // 512
constexpr int cM   = cB * cT;         // 8192 rows

#define WMMA_F16F32(a, b, c) \
  __builtin_amdgcn_wmma_f32_16x16x32_f16(false, (a), false, (b), (short)0, (c), false, false)

// ---------------------------------------------------------------------------
// Async global->LDS copy (CDNA5 GLOBAL_LOAD_ASYNC_TO_LDS_B128, ASYNCcnt).
// Builtin signature (from compiler diagnostics): (v4i* src_global, v4i* dst_lds,
// int offset, int cpol).  Guarded so the file compiles either way.
// ---------------------------------------------------------------------------
#if __has_builtin(__builtin_amdgcn_global_load_async_to_lds_b128)
#define HAS_ASYNC_CP 1
__device__ __forceinline__ void async_cp16(const half_t* g, half_t* l) {
  __builtin_amdgcn_global_load_async_to_lds_b128((v4i*)g, (v4i*)l, 0, 0);
}
#if __has_builtin(__builtin_amdgcn_s_wait_asynccnt)
#define WAIT_ASYNC(n) __builtin_amdgcn_s_wait_asynccnt(n)
#else
#define WAIT_ASYNC(n) asm volatile("s_wait_asynccnt %0" ::"i"(n) : "memory")
#endif
#else
#define HAS_ASYNC_CP 0
__device__ __forceinline__ void async_cp16(const half_t* g, half_t* l) {
  *(v8h*)l = *(const v8h*)g;           // sync fallback: global->reg->LDS
}
#define WAIT_ASYNC(n) ((void)0)
#endif

// ---------------------------------------------------------------------------
// RMSNorm: f32 row -> f16 row (h = g * x / sqrt(mean(x^2)+eps))
// ---------------------------------------------------------------------------
__global__ void __launch_bounds__(256) rmsnorm_f16_kernel(
    const float* __restrict__ x, const float* __restrict__ gamma,
    half_t* __restrict__ out, int Dd)
{
  const int row = blockIdx.x;
  const float* xr = x + (size_t)row * Dd;
  float ss = 0.f;
  for (int i = threadIdx.x; i < Dd; i += 256) { float v = xr[i]; ss += v * v; }
  __shared__ float red[256];
  red[threadIdx.x] = ss;
  __syncthreads();
  for (int s = 128; s > 0; s >>= 1) {
    if (threadIdx.x < s) red[threadIdx.x] += red[threadIdx.x + s];
    __syncthreads();
  }
  const float inv = rsqrtf(red[0] / (float)Dd + 1e-8f);
  half_t* orow = out + (size_t)row * Dd;
  for (int i = threadIdx.x; i < Dd; i += 256)
    orow[i] = (half_t)(gamma[i] * xr[i] * inv);
}

// ---------------------------------------------------------------------------
// Weight convert + transpose: W (K,N) f32  ->  Wt (N,K) f16
// ---------------------------------------------------------------------------
__global__ void __launch_bounds__(256) convert_transpose_kernel(
    const float* __restrict__ W, half_t* __restrict__ Wt, int K, int N)
{
  size_t i = (size_t)blockIdx.x * 256 + threadIdx.x;
  if (i >= (size_t)K * N) return;
  int n = (int)(i / K);
  int k = (int)(i % K);
  Wt[i] = (half_t)W[(size_t)k * N + n];
}

// ---------------------------------------------------------------------------
// WMMA GEMM, LDS-tiled + double-buffered async copies.
//   C[M,N] = A[M,K](f16) @ Bt[N,K](f16)^T + bias (+resid)
// Block: 256 thr = 8 waves in a 2x4 grid; block tile 64x256; wave tile 32x64
// (2x4 accumulators).  Per k-chunk (BK=32): A 64x32 (4KB) and B 256x32 (16KB)
// staged to LDS with GLOBAL_LOAD_ASYNC_TO_LDS_B128, double buffered (40KB).
// outhT layout: [(row/2048)*N + col][2048] + (row%2048)  (for V^T).
// ---------------------------------------------------------------------------
constexpr int BM = 64, BN = 256, BK = 32;

__global__ void __launch_bounds__(256) gemm_wmma_kernel(
    const half_t* __restrict__ A, const half_t* __restrict__ Bt,
    const float* __restrict__ bias, const float* __restrict__ resid,
    float* __restrict__ outf, half_t* __restrict__ outh,
    half_t* __restrict__ outhT, int M, int N, int K)
{
  __shared__ __align__(16) half_t Atile[2][BM * BK];   // 2 x 4KB
  __shared__ __align__(16) half_t Btile[2][BN * BK];   // 2 x 16KB

  const int tid  = threadIdx.x;
  const int lane = tid & 31, wid = tid >> 5;
  const int wm = wid >> 2, wn = wid & 3;               // wave position (2x4)
  const int hh = lane >> 4, ln = lane & 15;

  const int nBlkN = N / BN;
  const int row0 = (blockIdx.x / nBlkN) * BM;
  const int n0   = (blockIdx.x % nBlkN) * BN;

  // stage one k-chunk: 5 async b128 copies per thread (1 for A, 4 for B)
  auto stage = [&](int buf, int kc) {
    {
      const int r = tid >> 2, c = tid & 3;             // 64 rows x 4 chunks
      async_cp16(A + (size_t)(row0 + r) * K + kc + c * 8,
                 &Atile[buf][r * BK + c * 8]);
    }
#pragma unroll
    for (int j = 0; j < 4; ++j) {                      // 256 rows x 4 chunks
      const int id = tid + 256 * j;
      const int r = id >> 2, c = id & 3;
      async_cp16(Bt + (size_t)(n0 + r) * K + kc + c * 8,
                 &Btile[buf][r * BK + c * 8]);
    }
  };

  v8f acc[2][4];
#pragma unroll
  for (int t = 0; t < 2; ++t)
#pragma unroll
    for (int j = 0; j < 4; ++j) acc[t][j] = (v8f){};

  stage(0, 0);
  int buf = 0;
  for (int kc = 0; kc < K; kc += BK) {
    if (kc + BK < K) {
      stage(buf ^ 1, kc + BK);
      WAIT_ASYNC(5);        // in-order: <=5 outstanding => current batch done
    } else {
      WAIT_ASYNC(0);
    }
    __syncthreads();

    v16h af[2], bf[4];
#pragma unroll
    for (int t = 0; t < 2; ++t) {
      const half_t* pa = &Atile[buf][(wm * 32 + t * 16 + ln) * BK + 16 * hh];
      v8h lo = *(const v8h*)pa;
      v8h hi = *(const v8h*)(pa + 8);
#pragma unroll
      for (int i = 0; i < 8; ++i) { af[t][i] = lo[i]; af[t][8 + i] = hi[i]; }
    }
#pragma unroll
    for (int j = 0; j < 4; ++j)
      bf[j] = *(const v16h*)&Btile[buf][(wn * 64 + j * 16 + ln) * BK + 16 * hh];

#pragma unroll
    for (int t = 0; t < 2; ++t)
#pragma unroll
      for (int j = 0; j < 4; ++j)
        acc[t][j] = WMMA_F16F32(af[t], bf[j], acc[t][j]);

    __syncthreads();
    buf ^= 1;
  }

  const int colb = n0 + wn * 64 + ln;
#pragma unroll
  for (int j = 0; j < 4; ++j) {
    const int col = colb + j * 16;
    const float bv = bias ? bias[col] : 0.f;
#pragma unroll
    for (int t = 0; t < 2; ++t)
#pragma unroll
      for (int r = 0; r < 8; ++r) {
        const int row = row0 + wm * 32 + t * 16 + r + 8 * hh;
        float val = acc[t][j][r] + bv;
        if (resid) val += resid[(size_t)row * N + col];
        if (outf)  outf[(size_t)row * N + col] = val;
        if (outh)  outh[(size_t)row * N + col] = (half_t)val;
        if (outhT) outhT[(((size_t)(row >> 11) * N + col) << 11) + (row & 2047)] = (half_t)val;
      }
  }
}

// ---------------------------------------------------------------------------
// Flash attention.  One wave = one (head, 16-row query tile).
// ---------------------------------------------------------------------------
__device__ __forceinline__ float hmax16(float v) {
  v = fmaxf(v, __shfl_xor(v, 1, 32));
  v = fmaxf(v, __shfl_xor(v, 2, 32));
  v = fmaxf(v, __shfl_xor(v, 4, 32));
  v = fmaxf(v, __shfl_xor(v, 8, 32));
  return v;
}
__device__ __forceinline__ float hsum16(float v) {
  v += __shfl_xor(v, 1, 32);
  v += __shfl_xor(v, 2, 32);
  v += __shfl_xor(v, 4, 32);
  v += __shfl_xor(v, 8, 32);
  return v;
}

__global__ void __launch_bounds__(256) flash_attn_kernel(
    const half_t* __restrict__ q, const half_t* __restrict__ kmat,
    const half_t* __restrict__ vT, half_t* __restrict__ o)
{
  __shared__ __align__(16) _Float16 pbuf[8][16 * 32];

  const int wid  = threadIdx.x >> 5;
  const int lane = threadIdx.x & 31;
  const int w    = blockIdx.x * 8 + wid;
  if (w >= (cB * cG * cHPG) * (cT / 16)) return;

  const int head = w >> 7;
  const int mt   = w & 127;
  const int b    = head >> 4;
  const int g    = (head >> 2) & 3;
  const int p    = head & 3;
  const int t0   = mt * 16;
  const int hh   = lane >> 4, ln = lane & 15;
  const int qc   = (g * cHPG + p) * cHD;
  const float scale = 0.088388347648f;   // 1/sqrt(128)

  const half_t* qrow = q + ((size_t)(b * cT + t0 + ln)) * cD + qc;
  v16h qf[4];
#pragma unroll
  for (int kc = 0; kc < 4; ++kc) {
    const half_t* pa = qrow + kc * 32 + 16 * hh;
    v8h a0 = *(const v8h*)pa;
    v8h a1 = *(const v8h*)(pa + 8);
#pragma unroll
    for (int i = 0; i < 8; ++i) { qf[kc][i] = a0[i]; qf[kc][8 + i] = a1[i]; }
  }

  const half_t* kbase = kmat + (size_t)b * cT * cGHD + g * cHD;
  const half_t* vbase = vT + ((size_t)b * cGHD + g * cHD) * cT;

  float mrow[8], lrow[8];
  v8f of[8];
#pragma unroll
  for (int r = 0; r < 8; ++r) { mrow[r] = -1e30f; lrow[r] = 0.f; }
#pragma unroll
  for (int nc = 0; nc < 8; ++nc) of[nc] = (v8f){};

  _Float16* pb = &pbuf[wid][0];

  for (int s0 = 0; s0 < cT; s0 += 32) {
    v8f sA = {}, sB = {};
#pragma unroll
    for (int kc = 0; kc < 4; ++kc) {
      v16h kf0 = *(const v16h*)(kbase + (size_t)(s0 + ln) * cGHD + kc * 32 + 16 * hh);
      v16h kf1 = *(const v16h*)(kbase + (size_t)(s0 + 16 + ln) * cGHD + kc * 32 + 16 * hh);
      sA = WMMA_F16F32(qf[kc], kf0, sA);
      sB = WMMA_F16F32(qf[kc], kf1, sB);
    }

    float alpha[8];
#pragma unroll
    for (int r = 0; r < 8; ++r) {
      const float a0 = sA[r] * scale;
      const float a1 = sB[r] * scale;
      const float mx = hmax16(fmaxf(a0, a1));
      const float mn = fmaxf(mrow[r], mx);
      alpha[r] = __expf(mrow[r] - mn);
      mrow[r]  = mn;
      const float p0 = __expf(a0 - mn);
      const float p1 = __expf(a1 - mn);
      lrow[r] = lrow[r] * alpha[r] + hsum16(p0 + p1);
      pb[(r + 8 * hh) * 32 + ln]      = (_Float16)p0;
      pb[(r + 8 * hh) * 32 + 16 + ln] = (_Float16)p1;
    }

#pragma unroll
    for (int nc = 0; nc < 8; ++nc)
#pragma unroll
      for (int r = 0; r < 8; ++r) of[nc][r] *= alpha[r];

    const _Float16* pp = pb + ln * 32 + 16 * hh;
    v8h x0 = *(const v8h*)pp;
    v8h x1 = *(const v8h*)(pp + 8);
    v16h pf;
#pragma unroll
    for (int i = 0; i < 8; ++i) { pf[i] = x0[i]; pf[8 + i] = x1[i]; }

#pragma unroll
    for (int nc = 0; nc < 8; ++nc) {
      v16h vf = *(const v16h*)(vbase + (size_t)(nc * 16 + ln) * cT + s0 + 16 * hh);
      of[nc] = WMMA_F16F32(pf, vf, of[nc]);
    }
  }

  float inv[8];
#pragma unroll
  for (int r = 0; r < 8; ++r) inv[r] = 1.f / lrow[r];

  half_t* obase = o + (size_t)(b * cT + t0) * cD + qc;
#pragma unroll
  for (int nc = 0; nc < 8; ++nc)
#pragma unroll
    for (int r = 0; r < 8; ++r)
      obase[(size_t)(r + 8 * hh) * cD + nc * 16 + ln] = (half_t)(of[nc][r] * inv[r]);
}

// ---------------------------------------------------------------------------
// SwiGLU gate: u (M, 2D) f32 -> g (M, D) f16
// ---------------------------------------------------------------------------
__global__ void __launch_bounds__(256) swiglu_kernel(
    const float* __restrict__ u, half_t* __restrict__ g)
{
  size_t i = (size_t)blockIdx.x * 256 + threadIdx.x;
  if (i >= (size_t)cM * cD) return;
  size_t row = i / cD;
  int d = (int)(i % cD);
  float x1 = u[row * (2 * cD) + d];
  float x2 = u[row * (2 * cD) + cD + d];
  g[i] = (half_t)((x1 / (1.f + __expf(-x1))) * x2);
}

// ---------------------------------------------------------------------------
// launch
// ---------------------------------------------------------------------------
extern "C" void kernel_launch(void* const* d_in, const int* in_sizes, int n_in,
                              void* d_out, int out_size, void* d_ws, size_t ws_size,
                              hipStream_t stream) {
  const float* x  = (const float*)d_in[0];
  const float* g1 = (const float*)d_in[1];
  const float* Wq = (const float*)d_in[2];
  const float* bq = (const float*)d_in[3];
  const float* Wk = (const float*)d_in[4];
  const float* bk = (const float*)d_in[5];
  const float* Wv = (const float*)d_in[6];
  const float* bv = (const float*)d_in[7];
  const float* Wo = (const float*)d_in[8];
  const float* bo = (const float*)d_in[9];
  const float* g2 = (const float*)d_in[10];
  const float* w1 = (const float*)d_in[11];
  const float* b1 = (const float*)d_in[12];
  const float* w2 = (const float*)d_in[13];
  const float* b2 = (const float*)d_in[14];
  float* out = (float*)d_out;

  char* wp = (char*)d_ws;
  auto alloc = [&](size_t bytes) -> char* {
    char* p = wp;
    wp += (bytes + 255) & ~(size_t)255;
    return p;
  };
  half_t* hf16   = (half_t*)alloc((size_t)cM * cD * 2);
  half_t* wq_t   = (half_t*)alloc((size_t)cD * cD * 2);
  half_t* wk_t   = (half_t*)alloc((size_t)cGHD * cD * 2);
  half_t* wv_t   = (half_t*)alloc((size_t)cGHD * cD * 2);
  half_t* wo_t   = (half_t*)alloc((size_t)cD * cD * 2);
  half_t* w1_t   = (half_t*)alloc((size_t)(2 * cD) * cD * 2);
  half_t* w2_t   = (half_t*)alloc((size_t)cD * cD * 2);
  half_t* q16    = (half_t*)alloc((size_t)cM * cD * 2);
  half_t* k16    = (half_t*)alloc((size_t)cM * cGHD * 2);
  half_t* vT16   = (half_t*)alloc((size_t)cM * cGHD * 2);
  half_t* attn16 = (half_t*)alloc((size_t)cM * cD * 2);
  float*  xa     = (float*)alloc((size_t)cM * cD * 4);
  half_t* h216   = (half_t*)alloc((size_t)cM * cD * 2);
  float*  ubuf   = (float*)alloc((size_t)cM * 2 * cD * 4);
  half_t* gsw16  = (half_t*)alloc((size_t)cM * cD * 2);

  rmsnorm_f16_kernel<<<cM, 256, 0, stream>>>(x, g1, hf16, cD);

  auto conv = [&](const float* W, half_t* Wt, int K, int N) {
    size_t tot = (size_t)K * N;
    convert_transpose_kernel<<<(int)((tot + 255) / 256), 256, 0, stream>>>(W, Wt, K, N);
  };
  conv(Wq, wq_t, cD, cD);
  conv(Wk, wk_t, cD, cGHD);
  conv(Wv, wv_t, cD, cGHD);
  conv(Wo, wo_t, cD, cD);
  conv(w1, w1_t, cD, 2 * cD);
  conv(w2, w2_t, cD, cD);

  auto gemm = [&](const half_t* A, const half_t* Bt, const float* bias,
                  const float* resid, float* outf, half_t* outh, half_t* outhT,
                  int M, int N, int K) {
    int blocks = (M / BM) * (N / BN);
    gemm_wmma_kernel<<<blocks, 256, 0, stream>>>(
        A, Bt, bias, resid, outf, outh, outhT, M, N, K);
  };

  gemm(hf16, wq_t, bq, nullptr, nullptr, q16, nullptr, cM, cD, cD);
  gemm(hf16, wk_t, bk, nullptr, nullptr, k16, nullptr, cM, cGHD, cD);
  gemm(hf16, wv_t, bv, nullptr, nullptr, nullptr, vT16, cM, cGHD, cD);

  {
    int waves = (cB * cG * cHPG) * (cT / 16);   // 8192
    flash_attn_kernel<<<waves / 8, 256, 0, stream>>>(q16, k16, vT16, attn16);
  }

  gemm(attn16, wo_t, bo, x, xa, nullptr, nullptr, cM, cD, cD);
  rmsnorm_f16_kernel<<<cM, 256, 0, stream>>>(xa, g2, h216, cD);
  gemm(h216, w1_t, b1, nullptr, ubuf, nullptr, nullptr, cM, 2 * cD, cD);

  {
    size_t tot = (size_t)cM * cD;
    swiglu_kernel<<<(int)((tot + 255) / 256), 256, 0, stream>>>(ubuf, gsw16);
  }

  gemm(gsw16, w2_t, b2, xa, out, nullptr, nullptr, cM, cD, cD);
}
